// SelfAttnV3_22368189677670
// MI455X (gfx1250) — compile-verified, matrix operand order checked by reading, not run on hardware
//
#include <hip/hip_runtime.h>
#include <stdint.h>

#define DEV_INLINE __device__ __forceinline__

typedef __attribute__((ext_vector_type(16))) _Float16 v16h;
typedef __attribute__((ext_vector_type(8)))  _Float16 v8h;
typedef __attribute__((ext_vector_type(4)))  _Float16 v4h;
typedef __attribute__((ext_vector_type(8)))  float    v8f;

static constexpr int Bb = 8, Nn = 2048, Dd = 1024;

// ---------------------------------------------------------------------------
// CDNA5 async copy: 16B global -> LDS, tracked by ASYNCcnt (ISA 15.18.3,
// GLOBAL_LOAD_ASYNC_TO_LDS_B128).  lds = LDS byte address (= low 32 bits of
// the generic pointer per the aperture mapping), g = 64-bit global address.
// ---------------------------------------------------------------------------
DEV_INLINE void async_cp16(uint32_t lds, const void* g) {
  asm volatile("global_load_async_to_lds_b128 %0, %1, off"
               :: "v"(lds), "v"(g) : "memory");
}
DEV_INLINE void wait_async0() {
  asm volatile("s_wait_asynccnt 0x0" ::: "memory");
}
DEV_INLINE uint32_t lds_addr(const void* p) {
  return (uint32_t)(uintptr_t)p;  // generic LDS addr: addr[31:0] = LDS offset
}

// ---------------------------------------------------------------------------
// fp32 -> f16 convert (vectorized: float4 in, 4x half out)
// ---------------------------------------------------------------------------
__global__ __launch_bounds__(256) void k_cvt_f16(const float* __restrict__ in,
                                                 _Float16* __restrict__ out,
                                                 long n) {
  long i = ((long)blockIdx.x * 256 + threadIdx.x) * 4;
  if (i >= n) return;
  float4 f = *(const float4*)(in + i);
  v4h h;
  h[0] = (_Float16)f.x; h[1] = (_Float16)f.y;
  h[2] = (_Float16)f.z; h[3] = (_Float16)f.w;
  *(v4h*)(out + i) = h;
}

// ---------------------------------------------------------------------------
// Tiled transpose (+ optional fp32->f16 convert):  out[c][r] = in[r][c]
// ---------------------------------------------------------------------------
template <typename Tin>
__global__ __launch_bounds__(256) void k_transpose_f16(
    const Tin* __restrict__ in, _Float16* __restrict__ out,
    int ldIn, int ldOut, long sIn, long sOut) {
  __shared__ _Float16 tile[32][33];
  in  += (long)blockIdx.z * sIn;
  out += (long)blockIdx.z * sOut;
  const int r0 = blockIdx.y * 32, c0 = blockIdx.x * 32;
  const int tr = threadIdx.x >> 3;
  const int tc = (threadIdx.x & 7) * 4;
#pragma unroll
  for (int j = 0; j < 4; ++j)
    tile[tr][tc + j] = (_Float16)in[(long)(r0 + tr) * ldIn + c0 + tc + j];
  __syncthreads();
#pragma unroll
  for (int j = 0; j < 4; ++j)
    out[(long)(c0 + tr) * ldOut + r0 + tc + j] = tile[tc + j][tr];
}

// ---------------------------------------------------------------------------
// WMMA fragment load from LDS (two ds_load_b128 per ISA 7.12.2 layout).
// ---------------------------------------------------------------------------
DEV_INLINE v16h ld_frag16(const _Float16* p) {
  v8h lo = *(const v8h*)p;
  v8h hi = *(const v8h*)(p + 16);
  v16h r;
#pragma unroll
  for (int i = 0; i < 8; ++i) { r[i] = lo[i]; r[i + 8] = hi[i]; }
  return r;
}

// ---------------------------------------------------------------------------
// WMMA GEMM:  C[z] = alpha * A[z] x B[z]^T (+ bias)
//   A: [M x Kd] f16 row-major (lda);  B: [N x Kd] f16 row-major (ldb)
//   C: f32 or f16 (ldc); sA/sB/sC per-batch element strides.
// Block: 256 thr (8 waves), tile 128x256, BK=32.  Wave grid 2(m) x 4(n),
// each wave owns 64x64 = 4x4 WMMA accumulators (128 acc VGPRs).
// Double-buffered LDS; next tile staged via async global->LDS DMA
// (ASYNCcnt) fully overlapped with the 16-WMMA chain; 1 barrier/iter.
// ---------------------------------------------------------------------------
template <bool HAS_BIAS, bool OUT_F32>
__global__ __launch_bounds__(256) void k_gemm_wmma(
    const _Float16* __restrict__ A, const _Float16* __restrict__ B,
    const float* __restrict__ bias, void* __restrict__ Cout,
    int Kd, int lda, int ldb, int ldc,
    long sA, long sB, long sC, float alpha) {
  constexpr int BM = 128, BN = 256, BK = 32, LDK = BK + 8;
  __shared__ _Float16 As[2][BM][LDK];  // [buf][m][k]   (2 x 10 KB)
  __shared__ _Float16 Bs[2][BN][LDK];  // [buf][n][k]   (2 x 20 KB)

  const int tid  = threadIdx.x;
  const int lane = tid & 31;
  const int wid  = tid >> 5;
  const int wm   = (wid & 1) * 64;   // wave row offset in tile
  const int wn   = (wid >> 1) * 64;  // wave col offset in tile
  const int m0   = blockIdx.y * BM;
  const int n0   = blockIdx.x * BN;
  A += (long)blockIdx.z * sA;
  B += (long)blockIdx.z * sB;

  const int hs = lane >> 4;   // K half-select per ISA layout
  const int lm = lane & 15;   // row (A) / col (B) within 16x16 tile

  // staging: 16B chunks; A tile 512 chunks (2/thread), B tile 1024 (4/thread)
  const int ar = tid >> 2;          // 0..63
  const int ac = (tid & 3) * 8;     // 0,8,16,24
  const _Float16* gA0 = A + (long)(m0 + ar) * lda + ac;
  const _Float16* gA1 = A + (long)(m0 + ar + 64) * lda + ac;
  const _Float16* gB0 = B + (long)(n0 + ar) * ldb + ac;
  const _Float16* gB1 = B + (long)(n0 + ar + 64) * ldb + ac;
  const _Float16* gB2 = B + (long)(n0 + ar + 128) * ldb + ac;
  const _Float16* gB3 = B + (long)(n0 + ar + 192) * ldb + ac;

  uint32_t lA0[2], lA1[2], lB0[2], lB1[2], lB2[2], lB3[2];
#pragma unroll
  for (int b = 0; b < 2; ++b) {
    lA0[b] = lds_addr(&As[b][ar][ac]);
    lA1[b] = lds_addr(&As[b][ar + 64][ac]);
    lB0[b] = lds_addr(&Bs[b][ar][ac]);
    lB1[b] = lds_addr(&Bs[b][ar + 64][ac]);
    lB2[b] = lds_addr(&Bs[b][ar + 128][ac]);
    lB3[b] = lds_addr(&Bs[b][ar + 192][ac]);
  }

  v8f acc[4][4] = {};

  // prologue: async-stage K-tile 0 into buffer 0
  async_cp16(lA0[0], gA0);
  async_cp16(lA1[0], gA1);
  async_cp16(lB0[0], gB0);
  async_cp16(lB1[0], gB1);
  async_cp16(lB2[0], gB2);
  async_cp16(lB3[0], gB3);
  wait_async0();
  __syncthreads();

  int buf = 0;
  for (int k0 = 0;;) {
    const bool not_last = (k0 + BK) < Kd;
    if (not_last) {
      // DMA next K-tile into the other buffer; overlaps with WMMA below.
      const int nb = buf ^ 1;
      async_cp16(lA0[nb], gA0 + BK);
      async_cp16(lA1[nb], gA1 + BK);
      async_cp16(lB0[nb], gB0 + BK);
      async_cp16(lB1[nb], gB1 + BK);
      async_cp16(lB2[nb], gB2 + BK);
      async_cp16(lB3[nb], gB3 + BK);
    }

    v16h af[4], bf[4];
#pragma unroll
    for (int i = 0; i < 4; ++i)
      af[i] = ld_frag16(&As[buf][wm + i * 16 + lm][hs * 8]);
#pragma unroll
    for (int j = 0; j < 4; ++j)
      bf[j] = ld_frag16(&Bs[buf][wn + j * 16 + lm][hs * 8]);
#pragma unroll
    for (int i = 0; i < 4; ++i)
#pragma unroll
      for (int j = 0; j < 4; ++j)
        acc[i][j] = __builtin_amdgcn_wmma_f32_16x16x32_f16(
            false, af[i], false, bf[j], (short)0, acc[i][j], false, false);

    if (!not_last) break;
    wait_async0();     // our async writes to buf^1 are done
    __syncthreads();   // visible to all waves; all waves done reading buf
    buf ^= 1;
    k0 += BK;
    gA0 += BK; gA1 += BK; gB0 += BK; gB1 += BK; gB2 += BK; gB3 += BK;
  }

  // epilogue: C/D layout = VGPR r -> row (hs*8 + r), lane lm -> col
#pragma unroll
  for (int i = 0; i < 4; ++i) {
#pragma unroll
    for (int j = 0; j < 4; ++j) {
      const int col = n0 + wn + j * 16 + lm;
      const float bv = HAS_BIAS ? bias[col] : 0.0f;
#pragma unroll
      for (int r = 0; r < 8; ++r) {
        const int row = m0 + wm + i * 16 + hs * 8 + r;
        const float v = acc[i][j][r] * alpha + bv;
        if constexpr (OUT_F32) {
          float* C = (float*)Cout + (long)blockIdx.z * sC;
          C[(long)row * ldc + col] = v;
        } else {
          _Float16* C = (_Float16*)Cout + (long)blockIdx.z * sC;
          C[(long)row * ldc + col] = (_Float16)v;
        }
      }
    }
  }
}

// ---------------------------------------------------------------------------
// Fused double softmax with mask (one block per score row of length 2048):
//   w = softmax(s); v = where(mask==0, -1e20, w); P = softmax(v) -> f16
// Vectorized I/O (float4/int4/v8h), wave32 shfl_xor + 8-wave LDS reductions.
// ---------------------------------------------------------------------------
DEV_INLINE float wave_max(float x) {
#pragma unroll
  for (int o = 16; o; o >>= 1) x = fmaxf(x, __shfl_xor(x, o, 32));
  return x;
}
DEV_INLINE float wave_sum(float x) {
#pragma unroll
  for (int o = 16; o; o >>= 1) x += __shfl_xor(x, o, 32);
  return x;
}

__global__ __launch_bounds__(256) void k_softmax2(const float* __restrict__ S,
                                                  const int* __restrict__ mask,
                                                  _Float16* __restrict__ P) {
  __shared__ float red[8];
  const long row = blockIdx.x;
  const int tid = threadIdx.x, wid = tid >> 5, lane = tid & 31;
  const int base = tid * 8;
  const float* s  = S    + row * (long)Nn + base;
  const int*   mk = mask + row * (long)Nn + base;
  _Float16*    p  = P    + row * (long)Nn + base;

  float v[8];
  {
    float4 f0 = *(const float4*)s;
    float4 f1 = *(const float4*)(s + 4);
    v[0] = f0.x; v[1] = f0.y; v[2] = f0.z; v[3] = f0.w;
    v[4] = f1.x; v[5] = f1.y; v[6] = f1.z; v[7] = f1.w;
  }
  float mx = -3.0e38f;
#pragma unroll
  for (int i = 0; i < 8; ++i) mx = fmaxf(mx, v[i]);
  mx = wave_max(mx);
  if (lane == 0) red[wid] = mx;
  __syncthreads();
  float m1 = -3.0e38f;
#pragma unroll
  for (int i = 0; i < 8; ++i) m1 = fmaxf(m1, red[i]);
  __syncthreads();

  float zs = 0.f;
#pragma unroll
  for (int i = 0; i < 8; ++i) { v[i] = __expf(v[i] - m1); zs += v[i]; }
  zs = wave_sum(zs);
  if (lane == 0) red[wid] = zs;
  __syncthreads();
  float Z1 = 0.f;
#pragma unroll
  for (int i = 0; i < 8; ++i) Z1 += red[i];
  __syncthreads();
  const float inv1 = 1.0f / Z1;

  int m[8];
  {
    int4 i0 = *(const int4*)mk;
    int4 i1 = *(const int4*)(mk + 4);
    m[0] = i0.x; m[1] = i0.y; m[2] = i0.z; m[3] = i0.w;
    m[4] = i1.x; m[5] = i1.y; m[6] = i1.z; m[7] = i1.w;
  }
  float mx2 = -3.0e38f;
#pragma unroll
  for (int i = 0; i < 8; ++i) {
    float wv = v[i] * inv1;
    v[i] = (m[i] != 0) ? wv : -1.0e20f;
    mx2 = fmaxf(mx2, v[i]);
  }
  mx2 = wave_max(mx2);
  if (lane == 0) red[wid] = mx2;
  __syncthreads();
  float m2 = -3.0e38f;
#pragma unroll
  for (int i = 0; i < 8; ++i) m2 = fmaxf(m2, red[i]);
  __syncthreads();

  float z2 = 0.f;
#pragma unroll
  for (int i = 0; i < 8; ++i) { v[i] = __expf(v[i] - m2); z2 += v[i]; }
  z2 = wave_sum(z2);
  if (lane == 0) red[wid] = z2;
  __syncthreads();
  float Z2 = 0.f;
#pragma unroll
  for (int i = 0; i < 8; ++i) Z2 += red[i];
  const float inv2 = 1.0f / Z2;
  v8h ph;
#pragma unroll
  for (int i = 0; i < 8; ++i) ph[i] = (_Float16)(v[i] * inv2);
  *(v8h*)p = ph;
}

// ---------------------------------------------------------------------------
// Launcher
// ---------------------------------------------------------------------------
extern "C" void kernel_launch(void* const* d_in, const int* in_sizes, int n_in,
                              void* d_out, int out_size, void* d_ws,
                              size_t ws_size, hipStream_t stream) {
  (void)in_sizes; (void)n_in; (void)out_size; (void)ws_size;
  const float* x     = (const float*)d_in[0];  // [8,2048,1024]
  const int*   mask  = (const int*)d_in[1];    // [8,2048,2048]
  const float* W_qkv = (const float*)d_in[2];  // [1024,3072]
  const float* b_qkv = (const float*)d_in[3];  // [3072]
  const float* W_out = (const float*)d_in[4];  // [1024,1024]
  const float* b_out = (const float*)d_in[5];  // [1024]
  float* out = (float*)d_out;                  // [8,2048,1024] f32

  char* w = (char*)d_ws;
  auto take = [&](long bytes) { char* p = w; w += ((bytes + 255) & ~255L); return p; };
  _Float16* xh   = (_Float16*)take((long)Bb * Nn * Dd * 2);      //  32 MB
  _Float16* WqT  = (_Float16*)take((long)3 * Dd * Dd * 2);       //   6 MB  [3D][D]
  _Float16* WoT  = (_Float16*)take((long)Dd * Dd * 2);           //   2 MB  [D][D]
  _Float16* qkvh = (_Float16*)take((long)Bb * Nn * 3 * Dd * 2);  //  96 MB
  _Float16* Vt   = (_Float16*)take((long)Bb * Dd * Nn * 2);      //  32 MB  [b][d][n]
  float*    Smat = (float*)take((long)Bb * Nn * Nn * 4);         // 128 MB
  _Float16* P    = (_Float16*)take((long)Bb * Nn * Nn * 2);      //  64 MB
  _Float16* Oh   = (_Float16*)take((long)Bb * Nn * Dd * 2);      //  32 MB

  // 1) x -> f16; weights -> f16 transposed (so every GEMM is A*B^T)
  k_cvt_f16<<<(Bb * Nn * Dd) / 1024, 256, 0, stream>>>(x, xh, (long)Bb * Nn * Dd);
  k_transpose_f16<float><<<dim3(3 * Dd / 32, Dd / 32, 1), 256, 0, stream>>>(
      W_qkv, WqT, 3 * Dd, Dd, 0, 0);
  k_transpose_f16<float><<<dim3(Dd / 32, Dd / 32, 1), 256, 0, stream>>>(
      W_out, WoT, Dd, Dd, 0, 0);

  // 2) qkv[16384,3072] = xh @ WqT^T + b_qkv  (f16 out)
  k_gemm_wmma<true, false><<<dim3(3 * Dd / 256, Bb * Nn / 128, 1), 256, 0, stream>>>(
      xh, WqT, b_qkv, qkvh, Dd, Dd, Dd, 3 * Dd, 0, 0, 0, 1.0f);

  // 3) V -> Vt[b][d][n]  (f16 transpose; V = cols 2048..3071 of qkvh)
  k_transpose_f16<_Float16><<<dim3(Dd / 32, Nn / 32, Bb), 256, 0, stream>>>(
      qkvh + 2 * Dd, Vt, 3 * Dd, Nn, (long)Nn * 3 * Dd, (long)Dd * Nn);

  // 4) S[b] = Q[b] @ K[b]^T * (1/32)  per batch (f32 out)
  k_gemm_wmma<false, true><<<dim3(Nn / 256, Nn / 128, Bb), 256, 0, stream>>>(
      qkvh, qkvh + Dd, nullptr, Smat, Dd, 3 * Dd, 3 * Dd, Nn,
      (long)Nn * 3 * Dd, (long)Nn * 3 * Dd, (long)Nn * Nn, 0.03125f);

  // 5) P = softmax(mask(softmax(S)))   (f16 out)
  k_softmax2<<<Bb * Nn, 256, 0, stream>>>(Smat, mask, P);

  // 6) Oh[b] = P[b] @ Vt[b]^T  per batch (f16 out)
  k_gemm_wmma<false, false><<<dim3(Dd / 256, Nn / 128, Bb), 256, 0, stream>>>(
      P, Vt, nullptr, Oh, Nn, Nn, Nn, Dd,
      (long)Nn * Nn, (long)Dd * Nn, (long)Nn * Dd, 1.0f);

  // 7) out = Oh @ WoT^T + b_out  (f32 out)
  k_gemm_wmma<true, true><<<dim3(Dd / 256, Bb * Nn / 128, 1), 256, 0, stream>>>(
      Oh, WoT, b_out, out, Dd, Dd, Dd, Dd, 0, 0, 0, 1.0f);
}
